// SelectiveWKV_1B_5368709120020
// MI455X (gfx1250) — matrix operand order repaired and checked by reading, not compiled, not run on hardware
//
#include <hip/hip_runtime.h>
#include <hip/hip_bf16.h>

// ---------------------------------------------------------------------------
// SelectiveWKV (RWKV-style) for MI455X / gfx1250.
// GEMMs in bf16 via v_wmma_f32_16x16x32_bf16 (f32 accumulate).
// Each wave computes a 32x64 output tile: 8 WMMAs per k-step over 12 b128
// loads (1.5 loads/WMMA), B fragments reused across two A fragments.
// Scan in f32 with LDS staging + wave32 shfl reduction.
// ---------------------------------------------------------------------------

typedef __attribute__((ext_vector_type(16))) __bf16 v16bf;
typedef __attribute__((ext_vector_type(8)))  __bf16 v8bf;
typedef __attribute__((ext_vector_type(8)))  float  v8f;

#define DM   1024      // d_model
#define HS   64        // head size
#define NH   16        // heads
#define BB   4         // batch
#define TT   2048      // seq len
#define ROWS (BB*TT)   // 8192 GEMM rows

__device__ __forceinline__ float sigmoidf_(float x) {
    return 1.0f / (1.0f + __expf(-x));
}

// ---------------------------------------------------------------------------
// Weight transpose + f32 -> bf16 convert:  dst[n*K + k] = bf16(src[k*N + n])
// ---------------------------------------------------------------------------
__global__ __launch_bounds__(256) void transpose_cvt_bf16(const float* __restrict__ src,
                                                          __bf16* __restrict__ dst) {
    int gid = blockIdx.x * 256 + threadIdx.x;   // over 1024*1024
    int k = gid >> 10;          // row of src
    int n = gid & 1023;         // col of src
    dst[(size_t)n * DM + k] = (__bf16)src[gid];
}

// ---------------------------------------------------------------------------
// LayerNorm: one block per row, 256 threads, 4 elems/thread. Output bf16.
// ---------------------------------------------------------------------------
__global__ __launch_bounds__(256) void layernorm_bf16(const float* __restrict__ x,
                                                      const float* __restrict__ g,
                                                      const float* __restrict__ b,
                                                      __bf16* __restrict__ xn) {
    const int row = blockIdx.x;
    const float* xr = x + (size_t)row * DM;
    float v[4];
    float s = 0.f, ss = 0.f;
#pragma unroll
    for (int j = 0; j < 4; ++j) {
        v[j] = xr[threadIdx.x + 256 * j];
        s  += v[j];
        ss += v[j] * v[j];
    }
    // wave32 reduction
#pragma unroll
    for (int off = 16; off > 0; off >>= 1) {
        s  += __shfl_xor(s,  off, 32);
        ss += __shfl_xor(ss, off, 32);
    }
    __shared__ float rs[8], rss[8];
    if ((threadIdx.x & 31) == 0) {
        rs [threadIdx.x >> 5] = s;
        rss[threadIdx.x >> 5] = ss;
    }
    __syncthreads();
    float S = 0.f, SS = 0.f;
#pragma unroll
    for (int i = 0; i < 8; ++i) { S += rs[i]; SS += rss[i]; }
    const float mu  = S * (1.0f / DM);
    const float var = SS * (1.0f / DM) - mu * mu;
    const float inv = rsqrtf(var + 1e-5f);
    __bf16* o = xn + (size_t)row * DM;
#pragma unroll
    for (int j = 0; j < 4; ++j) {
        int c = threadIdx.x + 256 * j;
        o[c] = (__bf16)(((v[j] - mu) * inv) * g[c] + b[c]);
    }
}

// ---------------------------------------------------------------------------
// WMMA fragment loads (ISA 7.12.2, 16-bit layouts, wave32).
// A 16x32 (MxK), row-major A[M,K]:
//   lane<16 : row M=lane,     K = k0+[0..7] then k0+16+[0..7]
//   lane>=16: row M=lane-16,  K = k0+8+[0..7] then k0+24+[0..7]
// ---------------------------------------------------------------------------
__device__ __forceinline__ v16bf load_frag_a(const __bf16* __restrict__ base,
                                             int m0, int k0, int lane) {
    const int r  = lane & 15;
    const int kh = (lane >> 4) * 8;
    const __bf16* p = base + (size_t)(m0 + r) * DM + k0 + kh;
    v8bf lo = *(const v8bf*)(p);
    v8bf hi = *(const v8bf*)(p + 16);
    return __builtin_shufflevector(lo, hi, 0,1,2,3,4,5,6,7,8,9,10,11,12,13,14,15);
}

// B 32x16 (KxN) from W^T[N,K] row-major:
//   lane<16 : col N=n0+lane,    K = k0+[0..7] then k0+16+[0..7]
//   lane>=16: col N=n0+lane-16, K = k0+8+[0..7] then k0+24+[0..7]
__device__ __forceinline__ v16bf load_frag_b(const __bf16* __restrict__ wt,
                                             int n0, int k0, int lane) {
    const int n  = lane & 15;
    const int kh = (lane >> 4) * 8;
    const __bf16* p = wt + (size_t)(n0 + n) * DM + k0 + kh;
    v8bf lo = *(const v8bf*)(p);
    v8bf hi = *(const v8bf*)(p + 16);
    return __builtin_shufflevector(lo, hi, 0,1,2,3,4,5,6,7,8,9,10,11,12,13,14,15);
}

// ---------------------------------------------------------------------------
// GEMM  C[M,N] = A[M,K=1024] @ W (W given as bf16 W^T[N,K]),  M=8192, N=1024.
// Block = 256 threads = 8 waves; wave computes a 32(M)x64(N) tile
// (2 A frags x 4 B frags = 8 WMMAs per k-step; each B frag feeds 2 WMMAs).
// Block tile = 256(M) x 64(N). Grid = (8192/256)*(1024/64) = 512 blocks.
// EP: 0=none, 1=sigmoid(x+bias[col]), 2=sigmoid(x)
// ---------------------------------------------------------------------------
template <int EP>
__global__ __launch_bounds__(256) void gemm_wmma_bf16(const __bf16* __restrict__ A,
                                                      const __bf16* __restrict__ Wt,
                                                      const float* __restrict__ bias,
                                                      float* __restrict__ Cf,
                                                      __bf16* __restrict__ Cb) {
    const int lane   = threadIdx.x & 31;
    const int wave   = threadIdx.x >> 5;          // 0..7 : M slab inside block
    const int ntiles = DM / 64;                   // 16
    const int m0 = (blockIdx.x / ntiles) * 256 + wave * 32;   // wave: rows [m0, m0+32)
    const int n0 = (blockIdx.x % ntiles) * 64;

    v8f acc[2][4];
#pragma unroll
    for (int u = 0; u < 2; ++u)
#pragma unroll
        for (int j = 0; j < 4; ++j) acc[u][j] = (v8f){};

    for (int k0 = 0; k0 < DM; k0 += 32) {
        // prefetch next k-slab of A (gfx1250 global_prefetch path)
        if (k0 + 32 < DM)
            __builtin_prefetch(A + (size_t)(m0 + (lane & 15)) * DM + k0 + 32, 0, 1);

        v16bf a0 = load_frag_a(A, m0,      k0, lane);
        v16bf a1 = load_frag_a(A, m0 + 16, k0, lane);
#pragma unroll
        for (int j = 0; j < 4; ++j) {
            v16bf b = load_frag_b(Wt, n0 + j * 16, k0, lane);
            acc[0][j] = __builtin_amdgcn_wmma_f32_16x16x32_bf16(false, a0, false, b, (short)0, acc[0][j], false, false);
            acc[1][j] = __builtin_amdgcn_wmma_f32_16x16x32_bf16(false, a1, false, b, (short)0, acc[1][j], false, false);
        }
    }

    // C/D 16x16 f32 layout: lane<16 -> rows 0..7 (vgpr i = row i), col=lane;
    //                       lane>=16 -> rows 8..15, col = lane-16.
    const int cn = lane & 15;
#pragma unroll
    for (int u = 0; u < 2; ++u) {
        const int rbase = m0 + u * 16 + ((lane >> 4) ? 8 : 0);
#pragma unroll
        for (int j = 0; j < 4; ++j) {
            const int col = n0 + j * 16 + cn;
            const float bs = (EP == 1) ? bias[col] : 0.0f;
#pragma unroll
            for (int i = 0; i < 8; ++i) {
                float v = acc[u][j][i];
                if (EP == 1) v = sigmoidf_(v + bs);
                if (EP == 2) v = sigmoidf_(v);
                const size_t idx = (size_t)(rbase + i) * DM + col;
                if (Cf) Cf[idx] = v;
                if (Cb) Cb[idx] = (__bf16)v;
            }
        }
    }
}

// ---------------------------------------------------------------------------
// Sequential WKV scan. Grid = B*H = 64 blocks, 256 threads per block.
// Thread (dg = tid&3, e = tid>>2) owns state rows d in [dg*16, dg*16+16),
// column e. Per step: stage k/v/w/r vectors to LDS, update 16 state elems,
// reduce r.state across the 4 dg groups with wave32 shfl_xor.
// ---------------------------------------------------------------------------
__global__ __launch_bounds__(256) void wkv_scan(const float* __restrict__ kbuf,
                                                const float* __restrict__ vbuf,
                                                const float* __restrict__ wbuf,
                                                const float* __restrict__ rbuf,
                                                __bf16* __restrict__ outbf,
                                                float* __restrict__ state_out) {
    const int bh = blockIdx.x;            // 0..63
    const int b  = bh >> 4;
    const int h  = bh & 15;
    const int tid = threadIdx.x;
    const int dg  = tid & 3;
    const int e   = tid >> 2;

    float s[16];
#pragma unroll
    for (int i = 0; i < 16; ++i) s[i] = 0.0f;

    // stage[0]=k, stage[1]=v, stage[2]=w, stage[3]=r
    __shared__ float stage[4][HS];
    const float* srcs[4] = {kbuf, vbuf, wbuf, rbuf};  // runtime values: OK
    const int sel = tid >> 6;             // which of k/v/w/r this thread stages
    const int idx = tid & 63;

    const size_t base = (size_t)b * TT * DM + (size_t)h * HS;

    for (int t = 0; t < TT; ++t) {
        const size_t roff = base + (size_t)t * DM;
        stage[sel][idx] = srcs[sel][roff + idx];
        __syncthreads();

        const float vv = stage[1][e];
        float acc = 0.0f;
#pragma unroll
        for (int i = 0; i < 16; ++i) {
            const int d = dg * 16 + i;
            float sd = s[i];
            sd = sd - stage[2][d] * sd;            // (1 - w_d) * state
            sd = fmaf(stage[0][d], vv, sd);        // + k_d * v_e
            s[i] = sd;
            acc = fmaf(stage[3][d], sd, acc);      // r . state column
        }
        // dg groups sit in adjacent lanes (tid = dg + 4*e) -> wave32 xor reduce
        acc += __shfl_xor(acc, 1, 32);
        acc += __shfl_xor(acc, 2, 32);
        if (dg == 0) outbf[roff + e] = (__bf16)acc;
        __syncthreads();                           // protect LDS for next step
    }

    // final state [B,H,d,e]
#pragma unroll
    for (int i = 0; i < 16; ++i) {
        const int d = dg * 16 + i;
        state_out[((size_t)bh * HS + d) * HS + e] = s[i];
    }
}

// ---------------------------------------------------------------------------
// Host-side orchestration
// ---------------------------------------------------------------------------
extern "C" void kernel_launch(void* const* d_in, const int* in_sizes, int n_in,
                              void* d_out, int out_size, void* d_ws, size_t ws_size,
                              hipStream_t stream) {
    const float* x    = (const float*)d_in[0];
    const float* ln_g = (const float*)d_in[1];
    const float* ln_b = (const float*)d_in[2];
    const float* Wx   = (const float*)d_in[3];
    const float* Ww   = (const float*)d_in[4];
    const float* bw   = (const float*)d_in[5];
    const float* Wk   = (const float*)d_in[6];
    const float* Wv   = (const float*)d_in[7];
    const float* Wr   = (const float*)d_in[8];
    const float* Wo   = (const float*)d_in[9];

    // workspace carve-out
    size_t off = 0;
    auto carve = [&](size_t bytes) {
        void* p = (char*)d_ws + off;
        off += (bytes + 255) & ~(size_t)255;
        return p;
    };
    const size_t wbytes  = (size_t)DM * DM * sizeof(__bf16);     // 2 MB
    const size_t mbytesB = (size_t)ROWS * DM * sizeof(__bf16);   // 16 MB
    const size_t mbytesF = (size_t)ROWS * DM * sizeof(float);    // 32 MB

    __bf16* WxT = (__bf16*)carve(wbytes);
    __bf16* WwT = (__bf16*)carve(wbytes);
    __bf16* WkT = (__bf16*)carve(wbytes);
    __bf16* WvT = (__bf16*)carve(wbytes);
    __bf16* WrT = (__bf16*)carve(wbytes);
    __bf16* WoT = (__bf16*)carve(wbytes);
    __bf16* xn      = (__bf16*)carve(mbytesB);
    __bf16* tmpbf   = (__bf16*)carve(mbytesB);   // xn @ Wx (bf16)
    float*  wdec    = (float*)carve(mbytesF);    // sigmoid(tmp @ Ww + bw)
    float*  kproj   = (float*)carve(mbytesF);
    float*  vproj   = (float*)carve(mbytesF);
    float*  rgate   = (float*)carve(mbytesF);
    __bf16* scanout = (__bf16*)carve(mbytesB);

    float* y_out     = (float*)d_out;                       // [B,T,D]
    float* state_out = y_out + (size_t)ROWS * DM;           // [B,H,hs,hs]

    const dim3 blk(256);
    const dim3 gW(DM * DM / 256);             // 4096 blocks for weight transpose
    const dim3 gLN(ROWS);                     // 8192 rows
    const dim3 gG((ROWS / 256) * (DM / 64));  // 512 blocks per GEMM
    const dim3 gS(BB * NH);                   // 64 scan blocks

    // 1) weights -> bf16 transposed
    transpose_cvt_bf16<<<gW, blk, 0, stream>>>(Wx, WxT);
    transpose_cvt_bf16<<<gW, blk, 0, stream>>>(Ww, WwT);
    transpose_cvt_bf16<<<gW, blk, 0, stream>>>(Wk, WkT);
    transpose_cvt_bf16<<<gW, blk, 0, stream>>>(Wv, WvT);
    transpose_cvt_bf16<<<gW, blk, 0, stream>>>(Wr, WrT);
    transpose_cvt_bf16<<<gW, blk, 0, stream>>>(Wo, WoT);

    // 2) LayerNorm
    layernorm_bf16<<<gLN, blk, 0, stream>>>(x, ln_g, ln_b, xn);

    // 3..7) projections (WMMA bf16)
    gemm_wmma_bf16<0><<<gG, blk, 0, stream>>>(xn,    WxT, nullptr, nullptr, tmpbf);
    gemm_wmma_bf16<1><<<gG, blk, 0, stream>>>(tmpbf, WwT, bw,      wdec,    nullptr);
    gemm_wmma_bf16<0><<<gG, blk, 0, stream>>>(xn,    WkT, nullptr, kproj,   nullptr);
    gemm_wmma_bf16<0><<<gG, blk, 0, stream>>>(xn,    WvT, nullptr, vproj,   nullptr);
    gemm_wmma_bf16<2><<<gG, blk, 0, stream>>>(xn,    WrT, nullptr, rgate,   nullptr);

    // 8) sequential scan (writes final state into d_out tail)
    wkv_scan<<<gS, blk, 0, stream>>>(kproj, vproj, wdec, rgate, scanout, state_out);

    // 9) output projection
    gemm_wmma_bf16<0><<<gG, blk, 0, stream>>>(scanout, WoT, nullptr, y_out, nullptr);
}
